// MDN_RNN_56324201120387
// MI455X (gfx1250) — matrix operand order repaired and checked by reading, not compiled, not run on hardware
//
#include <hip/hip_runtime.h>

// ---------------------------------------------------------------------------
// MDN-RNN for MI455X (gfx1250, wave32, WMMA bf16 16x16x32, f32 accumulate)
//   S=2048 seq, B=64 batch, Z+A=35 in, H=256 hidden, 4H=1024 gates, M*Z=160
// Phase 0: pre-swizzle all weight B-fragments (bf16) into workspace.
// Phase 1: xg = X @ W_ih^T           (WMMA, TDM-fed LDS fragments)
// Phase 2: sequential LSTM, 8 persistent WGs, W_hh register-resident,
//          c-state in registers, h ping-pong via L2 + atomic step barrier.
// Phase 3: MDN heads + in-register softmax/exp (WMMA, TDM-fed LDS fragments)
// ---------------------------------------------------------------------------

#define S_LEN 2048
#define BATCH 64
#define HID   256
#define GATES 1024
#define NIN   35
#define MZ_   160

typedef __attribute__((ext_vector_type(16))) __bf16   v16bf;
typedef __attribute__((ext_vector_type(8)))  float    v8f;
typedef __attribute__((ext_vector_type(4)))  unsigned u32x4;
typedef __attribute__((ext_vector_type(2)))  unsigned u32x2;
typedef __attribute__((ext_vector_type(4)))  unsigned v4u;
typedef __attribute__((ext_vector_type(8)))  int      v8i;
typedef __attribute__((ext_vector_type(4)))  int      v4i;

union Frag  { v16bf v; u32x4 q[2]; };
union Frag4 { __bf16 h[4]; u32x2 q; };

#if defined(__has_builtin)
#if __has_builtin(__builtin_amdgcn_tensor_load_to_lds)
#define HAVE_TDM 1
#else
#define HAVE_TDM 0
#endif
#else
#define HAVE_TDM 0
#endif

__device__ __forceinline__ v8f wmma_bf16(v16bf a, v16bf b, v8f c) {
  return __builtin_amdgcn_wmma_f32_16x16x32_bf16(
      false, a, false, b, (short)0, c, false, false);
}

__device__ __forceinline__ float sigf(float x) { return 1.f / (1.f + __expf(-x)); }
__device__ __forceinline__ float tanhf_(float x) {
  float e = __expf(-2.f * fabsf(x));
  float t = (1.f - e) / (1.f + e);
  return copysignf(t, x);
}

#if HAVE_TDM
// 1-D contiguous DMA: global -> LDS, nbytes must be a multiple of 8.
// D# per CDNA5 ISA (2 descriptor groups used; groups
// 2/3 and the trailing group are zero => tensor limited to 2-D, which is all
// we need for a flat copy).  This toolchain exposes the 6-arg builtin:
//   (v4u g0, v8i g1, v4i g2, v4i g3, v8i g4, i32 cpol)
__device__ __forceinline__ void tdm_copy_to_lds(unsigned lds_off, const void* src,
                                                unsigned nbytes) {
  unsigned long long ga = (unsigned long long)(size_t)src;
  unsigned n8 = nbytes >> 3;                       // 8-byte elements
  v4u g0;
  g0[0] = 1u;                                      // count=1, user mode
  g0[1] = lds_off;                                 // lds_addr
  g0[2] = (unsigned)(ga & 0xffffffffu);            // global_addr[31:0]
  g0[3] = (unsigned)((ga >> 32) & 0x01ffffffu)     // global_addr[56:32]
        | (2u << 30);                              // type = 2 ("image")
  v8i g1;
  g1[0] = (int)(3u << 16);                         // wg_mask=0, data_size=8B
  g1[1] = (int)(n8 << 16);                         // tensor_dim0[15:0]
  g1[2] = (int)(1u << 16);                         // dim0[31:16]=0, tensor_dim1=1
  g1[3] = (int)(n8 << 16);                         // tile_dim0 = n8
  g1[4] = 0;                                       // tile_dim1/2 unused
  g1[5] = (int)n8;                                 // tensor_dim0_stride[31:0]
  g1[6] = 0;
  g1[7] = 0;
  v4i z4 = {0, 0, 0, 0};
  v8i z8 = {0, 0, 0, 0, 0, 0, 0, 0};
  __builtin_amdgcn_tensor_load_to_lds(g0, g1, z4, z4, z8, 0);
}
#endif

// ---------------------------------------------------------------------------
// Phase 0: pre-swizzle every WMMA B-fragment into workspace (bf16).
//   frag = 32 lanes x 32B (lane%16 = N, 16 K per lane in table order: 8h+s | 16+8h+s)
//   f <  128: W_ih   frags, f = nt*2 + ks          (nt 0..63, K padded to 64)
//   f <  640: W_hh   frags, id = ((dq*4+g)*8 + ks) (dq = hidden-dim/16)
//   f <  880: head   frags, id = head*80 + nt*8+ks
// ---------------------------------------------------------------------------
__global__ __launch_bounds__(32)
void pack_frags(const float* __restrict__ Wih, const float* __restrict__ Whh,
                const float* __restrict__ Wpi, const float* __restrict__ Wsig,
                const float* __restrict__ Wmu,
                __bf16* __restrict__ fragIH, __bf16* __restrict__ fragHH,
                __bf16* __restrict__ fragHD) {
  const int f = blockIdx.x, lane = threadIdx.x;
  Frag u;
  __bf16* dst;
  if (f < 128) {
    int ntl = f >> 1, ks = f & 1;
    int kb = ks * 32 + 8 * (lane >> 4);
#pragma unroll
    for (int e = 0; e < 16; ++e) {
      int n = ntl * 16 + (lane & 15);
      int k = kb + (e & 7) + ((e >> 3) * 16);
      u.v[e] = (__bf16)((k < NIN) ? Wih[(size_t)n * NIN + k] : 0.f);
    }
    dst = fragIH + (size_t)f * 512;
  } else if (f < 640) {
    int id = f - 128;
    int ks = id & 7, g = (id >> 3) & 3, dq = id >> 5;
    int kb = ks * 32 + 8 * (lane >> 4);
    int nb = g * 256 + dq * 16;
#pragma unroll
    for (int e = 0; e < 16; ++e)
      u.v[e] = (__bf16)Whh[(size_t)(nb + (lane & 15)) * HID + kb + (e & 7) + ((e >> 3) * 16)];
    dst = fragHH + (size_t)id * 512;
  } else {
    int id = f - 640;
    int head = id / 80, rem = id % 80;
    int nt = rem >> 3, ks = rem & 7;
    const float* W = head == 0 ? Wpi : (head == 1 ? Wsig : Wmu);
    int kb = ks * 32 + 8 * (lane >> 4);
#pragma unroll
    for (int e = 0; e < 16; ++e)
      u.v[e] = (__bf16)W[(size_t)(nt * 16 + (lane & 15)) * HID + kb + (e & 7) + ((e >> 3) * 16)];
    dst = fragHD + (size_t)id * 512;
  }
  u32x4* d = (u32x4*)dst;
  d[lane * 2 + 0] = u.q[0];
  d[lane * 2 + 1] = u.q[1];
}

// ---------------------------------------------------------------------------
// Phase 1: xg = concat(z, action) @ W_ih^T, stored bf16 gate-interleaved:
//   xg[row*1024 + d*4 + g]   (row = s*B + b, d = hidden dim, g = gate)
// grid (1024, 2) x 256 thr; y = 512-column half. 64KB of fragments TDM-loaded.
// ---------------------------------------------------------------------------
__global__ __launch_bounds__(256)
void xg_proj(const float* __restrict__ z, const float* __restrict__ act,
             const __bf16* __restrict__ fragIH, __bf16* __restrict__ xg) {
  extern __shared__ u32x4 lds[];                 // 64 frags * 64 u32x4
  const int wave = threadIdx.x >> 5, lane = threadIdx.x & 31;
  const int ntbase = blockIdx.y * 32;

#if HAVE_TDM
  if (threadIdx.x == 0) {
    tdm_copy_to_lds((unsigned)(size_t)lds,
                    fragIH + (size_t)blockIdx.y * (64 * 512), 64 * 1024);
    __builtin_amdgcn_s_wait_tensorcnt(0);
  }
#else
  {
    const u32x4* src = (const u32x4*)(fragIH + (size_t)blockIdx.y * (64 * 512));
    for (int i = threadIdx.x; i < 4096; i += 256) lds[i] = src[i];
  }
#endif
  __syncthreads();

  const int row0 = (blockIdx.x * 8 + wave) * 16;
  const int arow = row0 + (lane & 15);

  Frag A[2];
#pragma unroll
  for (int ks = 0; ks < 2; ++ks)
#pragma unroll
    for (int e = 0; e < 16; ++e) {
      int k = ks * 32 + 8 * (lane >> 4) + (e & 7) + ((e >> 3) * 16);
      float v = 0.f;
      if (k < 32)       v = z[(size_t)arow * 32 + k];
      else if (k < NIN) v = act[(size_t)arow * 3 + (k - 32)];
      A[ks].v[e] = (__bf16)v;
    }

  for (int ntl = 0; ntl < 32; ++ntl) {
    v8f acc = {0.f, 0.f, 0.f, 0.f, 0.f, 0.f, 0.f, 0.f};
#pragma unroll
    for (int ks = 0; ks < 2; ++ks) {
      Frag b;
      int f = ntl * 2 + ks;
      b.q[0] = lds[(f * 32 + lane) * 2 + 0];
      b.q[1] = lds[(f * 32 + lane) * 2 + 1];
      acc = wmma_bf16(A[ks].v, b.v, acc);
    }
    int col = (ntbase + ntl) * 16 + (lane & 15);
    int g = col >> 8, d = col & 255;               // gate-interleaved store
#pragma unroll
    for (int r = 0; r < 8; ++r) {
      size_t row = (size_t)row0 + r + 8 * (lane >> 4);
      xg[row * GATES + d * 4 + g] = (__bf16)acc[r];
    }
  }
}

// ---------------------------------------------------------------------------
// Phase 2: sequential LSTM (the latency-critical recurrence).
// ---------------------------------------------------------------------------
__global__ __launch_bounds__(256)
void lstm_seq(const __bf16* __restrict__ xg, const __bf16* __restrict__ fragHH,
              const float* __restrict__ bih, const float* __restrict__ bhh,
              __bf16* __restrict__ hbuf, float* __restrict__ hs,
              unsigned* __restrict__ ctr) {
  const int wg   = blockIdx.x;                 // 0..7 -> h-dims [wg*32, +32)
  const int wave = threadIdx.x >> 5;           // 0..7
  const int lane = threadIdx.x & 31;
  const int mi   = wave >> 1;                  // batch tile 0..3
  const int dq   = wg * 2 + (wave & 1);        // hidden-dim/16 tile 0..15
  const int d0   = dq * 16;

  // W_hh slice: 32 pre-swizzled fragments -> 256 register-resident VGPRs
  v16bf Bf[4][8];
  {
    const u32x4* fb = (const u32x4*)fragHH;
#pragma unroll
    for (int g = 0; g < 4; ++g)
#pragma unroll
      for (int ks = 0; ks < 8; ++ks) {
        int fid = (dq * 4 + g) * 8 + ks;
        Frag u;
        u.q[0] = fb[(fid * 32 + lane) * 2 + 0];
        u.q[1] = fb[(fid * 32 + lane) * 2 + 1];
        Bf[g][ks] = u.v;
      }
  }

  float biasv[4];
#pragma unroll
  for (int g = 0; g < 4; ++g) {
    int n = g * 256 + d0 + (lane & 15);
    biasv[g] = bih[n] + bhh[n];
  }

  float cst[8];
#pragma unroll
  for (int r = 0; r < 8; ++r) cst[r] = 0.f;

  const int brow = mi * 16 + (lane & 15);      // A-matrix row (batch)
  const int dd   = d0 + (lane & 15);           // hidden dim (C/D column)

  for (int t = 0; t < S_LEN; ++t) {
    const __bf16* hin  = hbuf + (size_t)(t & 1) * (BATCH * HID);
    __bf16*       hout = hbuf + (size_t)((t + 1) & 1) * (BATCH * HID);

    v8f acc[4];
#pragma unroll
    for (int g = 0; g < 4; ++g)
#pragma unroll
      for (int r = 0; r < 8; ++r) acc[g][r] = biasv[g];   // C init = bias

#pragma unroll
    for (int ks = 0; ks < 8; ++ks) {
      Frag a;   // A 16x32 bf16: two aligned 16B chunks of h(t)
      const u32x4* p =
          (const u32x4*)(hin + (size_t)brow * HID + ks * 32 + 8 * (lane >> 4));
      a.q[0] = p[0];
      a.q[1] = p[2];
#pragma unroll
      for (int g = 0; g < 4; ++g) acc[g] = wmma_bf16(a.v, Bf[g][ks], acc[g]);
    }

    // Prefetch next timestep's xg rows while the cell math runs.
    const __bf16* xgt = xg + (size_t)t * (BATCH * GATES);
    if (t + 1 < S_LEN)
      __builtin_prefetch(xgt + (BATCH * GATES) +
                         (size_t)(mi * 16 + 8 * (lane >> 4)) * GATES + dd * 4, 0, 3);

    // LSTM cell: gate-interleaved xg -> one 8B load per row.
#pragma unroll
    for (int r = 0; r < 8; ++r) {
      int b = mi * 16 + r + 8 * (lane >> 4);
      Frag4 x;
      x.q = *(const u32x2*)(xgt + (size_t)b * GATES + dd * 4);
      float iv = acc[0][r] + (float)x.h[0];
      float fv = acc[1][r] + (float)x.h[1];
      float gv = acc[2][r] + (float)x.h[2];
      float ov = acc[3][r] + (float)x.h[3];
      float cn = sigf(fv) * cst[r] + sigf(iv) * tanhf_(gv);
      cst[r] = cn;
      float hv = sigf(ov) * tanhf_(cn);
      hout[(size_t)b * HID + dd] = (__bf16)hv;
      hs[((size_t)t * BATCH + b) * HID + dd] = hv;
    }

    // Device-scope step barrier across the 8 workgroups.
    __threadfence();
    __syncthreads();
    if (threadIdx.x == 0) {
      __hip_atomic_fetch_add(ctr, 1u, __ATOMIC_ACQ_REL, __HIP_MEMORY_SCOPE_AGENT);
      unsigned target = 8u * (unsigned)(t + 1);
      while (__hip_atomic_load(ctr, __ATOMIC_ACQUIRE, __HIP_MEMORY_SCOPE_AGENT) < target)
        __builtin_amdgcn_s_sleep(1);
    }
    __syncthreads();
  }
}

// ---------------------------------------------------------------------------
// Phase 3: MDN heads. grid (1024, 3): y = head (pi | sigma | mu).
// 80KB of pre-swizzled fragments TDM-loaded to LDS per block.
// ---------------------------------------------------------------------------
__global__ __launch_bounds__(256)
void mdn_heads(const float* __restrict__ hs, const __bf16* __restrict__ fragHD,
               const float* __restrict__ bpi, const float* __restrict__ bsig,
               const float* __restrict__ bmu, float* __restrict__ out) {
  extern __shared__ u32x4 lds[];               // 80 frags * 64 u32x4
  const int wave = threadIdx.x >> 5, lane = threadIdx.x & 31;
  const int head = blockIdx.y;
  const float* bb = head == 0 ? bpi : (head == 1 ? bsig : bmu);

#if HAVE_TDM
  if (threadIdx.x == 0) {
    tdm_copy_to_lds((unsigned)(size_t)lds,
                    fragHD + (size_t)head * (80 * 512), 80 * 1024);
    __builtin_amdgcn_s_wait_tensorcnt(0);
  }
#else
  {
    const u32x4* src = (const u32x4*)(fragHD + (size_t)head * (80 * 512));
    for (int i = threadIdx.x; i < 5120; i += 256) lds[i] = src[i];
  }
#endif
  __syncthreads();

  const int row0 = (blockIdx.x * 8 + wave) * 16;
  float biasv[10];
#pragma unroll
  for (int nt = 0; nt < 10; ++nt) biasv[nt] = bb[nt * 16 + (lane & 15)];

  v8f acc[10];
#pragma unroll
  for (int nt = 0; nt < 10; ++nt)
#pragma unroll
    for (int r = 0; r < 8; ++r) acc[nt][r] = biasv[nt];

  const int arow = row0 + (lane & 15);
#pragma unroll
  for (int ks = 0; ks < 8; ++ks) {
    Frag a;
    const float* p = hs + (size_t)arow * HID + ks * 32 + 8 * (lane >> 4);
#pragma unroll
    for (int e = 0; e < 8; ++e) {
      a.v[e]     = (__bf16)p[e];
      a.v[8 + e] = (__bf16)p[16 + e];
    }
#pragma unroll
    for (int nt = 0; nt < 10; ++nt) {
      Frag b;
      int f = nt * 8 + ks;
      b.q[0] = lds[(f * 32 + lane) * 2 + 0];
      b.q[1] = lds[(f * 32 + lane) * 2 + 1];
      acc[nt] = wmma_bf16(a.v, b.v, acc[nt]);
    }
  }

  const size_t OFF_SIG = 20971520u, OFF_MU = 41943040u;
#pragma unroll
  for (int r = 0; r < 8; ++r) {
    size_t row = (size_t)row0 + r + 8 * (lane >> 4);
    if (head == 0) {                           // pi: softmax over M, / TEMP
#pragma unroll
      for (int zhi = 0; zhi < 2; ++zhi) {
        float v[5];
#pragma unroll
        for (int m = 0; m < 5; ++m) v[m] = acc[2 * m + zhi][r];
        float mx = v[0];
#pragma unroll
        for (int m = 1; m < 5; ++m) mx = fmaxf(mx, v[m]);
        float e[5], s = 0.f;
#pragma unroll
        for (int m = 0; m < 5; ++m) { e[m] = __expf(v[m] - mx); s += e[m]; }
        float inv = 1.f / (s * 1.3f);
#pragma unroll
        for (int m = 0; m < 5; ++m)
          out[row * MZ_ + m * 32 + zhi * 16 + (lane & 15)] = e[m] * inv;
      }
    } else {
      float* o = out + (head == 1 ? OFF_SIG : OFF_MU);
#pragma unroll
      for (int nt = 0; nt < 10; ++nt) {
        int col = nt * 16 + (lane & 15);
        float v = acc[nt][r];
        o[row * MZ_ + col] = (head == 1) ? __expf(v) : v;
      }
    }
  }
}

// ---------------------------------------------------------------------------
extern "C" void kernel_launch(void* const* d_in, const int* in_sizes, int n_in,
                              void* d_out, int out_size, void* d_ws, size_t ws_size,
                              hipStream_t stream) {
  (void)in_sizes; (void)n_in; (void)out_size; (void)ws_size;
  const float* z    = (const float*)d_in[0];
  const float* act  = (const float*)d_in[1];
  const float* Wih  = (const float*)d_in[2];
  const float* Whh  = (const float*)d_in[3];
  const float* bih  = (const float*)d_in[4];
  const float* bhh  = (const float*)d_in[5];
  const float* Wpi  = (const float*)d_in[6];
  const float* bpi  = (const float*)d_in[7];
  const float* Wsig = (const float*)d_in[8];
  const float* bsig = (const float*)d_in[9];
  const float* Wmu  = (const float*)d_in[10];
  const float* bmu  = (const float*)d_in[11];
  float* out = (float*)d_out;

  // Workspace: xg (bf16, 256MiB) | h ping-pong (64KiB) | ctr (256B) |
  //            fragIH (128KiB) | fragHH (512KiB) | fragHD (240KiB)
  const size_t XG_BYTES = (size_t)S_LEN * BATCH * GATES * 2;
  char* ws = (char*)d_ws;
  __bf16*   xg     = (__bf16*)ws;
  __bf16*   hbuf   = (__bf16*)(ws + XG_BYTES);
  unsigned* ctr    = (unsigned*)(ws + XG_BYTES + 65536);
  __bf16*   fragIH = (__bf16*)(ws + XG_BYTES + 65536 + 256);
  __bf16*   fragHH = (__bf16*)(ws + XG_BYTES + 65536 + 256 + 131072);
  __bf16*   fragHD = (__bf16*)(ws + XG_BYTES + 65536 + 256 + 131072 + 524288);

  // Reset h(0)=0 (both ping-pong buffers) + barrier counter every launch.
  (void)hipMemsetAsync(hbuf, 0, 65536 + 256, stream);

  float* hs = out + 62914560u;   // 4th output region: (S, B, H) f32

  pack_frags<<<dim3(880),     32,  0,         stream>>>(Wih, Whh, Wpi, Wsig, Wmu,
                                                        fragIH, fragHH, fragHD);
  xg_proj   <<<dim3(1024, 2), 256, 64 * 1024, stream>>>(z, act, fragIH, xg);
  lstm_seq  <<<dim3(8),       256, 0,         stream>>>(xg, fragHH, bih, bhh,
                                                        hbuf, hs, ctr);
  mdn_heads <<<dim3(1024, 3), 256, 80 * 1024, stream>>>(hs, fragHD,
                                                        bpi, bsig, bmu, out);
}
